// GATLayer_30769145708781
// MI455X (gfx1250) — matrix operand order repaired and checked by reading, not compiled, not run on hardware
//
#include <hip/hip_runtime.h>
#include <hip/hip_bf16.h>

// ---------------------------------------------------------------------------
// GAT layer for MI455X (gfx1250, wave32, WMMA + async-LDS staging).
// Shapes fixed by the reference: B=8, N=1024, C_IN=C_OUT=256, R=4, H=4, C=64.
// ---------------------------------------------------------------------------

typedef __attribute__((ext_vector_type(16))) __bf16 v16bf;
typedef __attribute__((ext_vector_type(8)))  float  v8f;

union Frag16 {           // one WMMA 16-bf16 operand = 32 bytes = two uint4
    v16bf v;
    uint4 q[2];
};

#define GAT_NEG_INF (-9.0e15f)
#define GAT_ALPHA   (0.2f)

__device__ __forceinline__ float leaky(float x) {
    return x > 0.0f ? x : GAT_ALPHA * x;
}

// ---- CDNA5 async global->LDS copy (ASYNCcnt path), with sync fallback -----
#if __has_builtin(__builtin_amdgcn_global_load_async_to_lds_b128) && \
    __has_builtin(__builtin_amdgcn_s_wait_asynccnt)
#define GAT_ASYNC_LDS 1
#else
#define GAT_ASYNC_LDS 0
#endif

#if GAT_ASYNC_LDS
typedef __attribute__((ext_vector_type(4))) int i32x4;
typedef __attribute__((address_space(1))) i32x4 as1_i32x4;
typedef __attribute__((address_space(3))) i32x4 as3_i32x4;
__device__ __forceinline__ void async_copy16(const void* g, void* l) {
    // generic->as(1): identical bit pattern; generic->as(3): LDS offset in [31:0]
    __builtin_amdgcn_global_load_async_to_lds_b128(
        (as1_i32x4*)(uintptr_t)g, (as3_i32x4*)(uint32_t)(uintptr_t)l, 0, 0);
}
#endif

// ------------------------------------------------------------------ prep ----
__global__ __launch_bounds__(256) void k_prep_nf(const float* __restrict__ in,
                                                 __bf16* __restrict__ outp) {
    int i = blockIdx.x * 256 + threadIdx.x;          // 8*1024*256 = 2M elems
    outp[i] = (__bf16)in[i];
}

// Wt[r][n][k] = W[r][k][n]  (bf16, K-contiguous per output column)
__global__ __launch_bounds__(256) void k_prep_w(const float* __restrict__ W,
                                                __bf16* __restrict__ wt) {
    int idx = blockIdx.x * 256 + threadIdx.x;        // 4*256*256 = 256K elems
    int r = idx >> 16, rem = idx & 65535;
    int n = rem >> 8, k = rem & 255;
    wt[idx] = (__bf16)W[(r * 256 + k) * 256 + n];
}

// ---------------------------------------------------------- feature GEMM ----
// One wave per (16-row tile of B*N, head(64 cols), r).  f32 accumulate via
// v_wmma_f32_16x16x32_bf16, then LDS-transpose epilogue writing
// f_t[r][b][h][c(64)][j(1024)] in bf16 (B-operand-friendly layout for pass B).
__global__ __launch_bounds__(32) void k_gemm_f(const __bf16* __restrict__ nf,
                                               const __bf16* __restrict__ wt,
                                               __bf16* __restrict__ ft) {
    const int mt = blockIdx.x;        // 0..511
    const int h  = blockIdx.y;        // 0..3   (64-column tile == head)
    const int r  = blockIdx.z;        // 0..3
    const int l  = threadIdx.x;
    const int lr = l & 15;
    const bool hih = l >= 16;

    const int m0  = mt * 16;          // global row in [0, 8192)
    const int b   = m0 >> 10;
    const int j0g = m0 & 1023;        // row within batch

    // A operand: lane holds row (l&15); K = {o..o+7, o+16..o+23}, o = hih?8:0
    const __bf16* ap = nf + (m0 + lr) * 256 + (hih ? 8 : 0);
    // B operand: lane holds col (l&15); K = {o..o+7, o+16..o+23}, o = hih?8:0
    const __bf16* wp0 = wt + (r * 256 + h * 64 + lr) * 256 + (hih ? 8 : 0);

    v8f acc[4] = {v8f{}, v8f{}, v8f{}, v8f{}};
    for (int k0 = 0; k0 < 256; k0 += 32) {
        Frag16 a;
        a.q[0] = *(const uint4*)(ap + k0);
        a.q[1] = *(const uint4*)(ap + k0 + 16);
#pragma unroll
        for (int ct = 0; ct < 4; ++ct) {
            Frag16 bf;
            const __bf16* wp = wp0 + ct * 16 * 256 + k0;
            bf.q[0] = *(const uint4*)(wp);
            bf.q[1] = *(const uint4*)(wp + 16);
            acc[ct] = __builtin_amdgcn_wmma_f32_16x16x32_bf16(
                false, a.v, false, bf.v, (short)0, acc[ct], false, false);
        }
    }

    // D layout: lane l, vgpr vi -> row (vi + (hih?8:0)), col (l&15).
    __shared__ __align__(16) __bf16 sh[64][16];   // [c_local][j_local]
#pragma unroll
    for (int ct = 0; ct < 4; ++ct)
#pragma unroll
        for (int vi = 0; vi < 8; ++vi)
            sh[ct * 16 + lr][vi + (hih ? 8 : 0)] = (__bf16)acc[ct][vi];
    __syncthreads();

    __bf16* fb = ft + (((r * 8 + b) * 4 + h) << 16);   // *65536
#pragma unroll
    for (int it = 0; it < 4; ++it) {
        int idx = it * 32 + l;            // 0..127 -> (c, j-half)
        int c = idx >> 1;
        int jh = (idx & 1) * 8;
        *(uint4*)(fb + c * 1024 + j0g + jh) = *(const uint4*)(&sh[c][jh]);
    }
}

// ------------------------------------------------------- li / lj vectors ----
// li[b][n][r*4+h] = sum_c f[b,n,h,c] * a_l[r,h,c];  lj likewise with a_r.
__global__ __launch_bounds__(256) void k_lilj(const __bf16* __restrict__ ft,
                                              const float* __restrict__ a,
                                              float* __restrict__ li,
                                              float* __restrict__ lj) {
    int n = blockIdx.x * 256 + threadIdx.x;    // 0..8191 over (b, j)
    int b = n >> 10, j = n & 1023;
#pragma unroll
    for (int r = 0; r < 4; ++r)
#pragma unroll
        for (int h = 0; h < 4; ++h) {
            const __bf16* f = ft + (((r * 8 + b) * 4 + h) << 16) + j;
            const float* av = a + (r * 4 + h) * 128;
            float sl = 0.f, sr = 0.f;
            for (int c = 0; c < 64; ++c) {
                float fv = (float)f[c * 1024];     // lane-coalesced (stride j)
                sl += fv * av[c];
                sr += fv * av[64 + c];
            }
            li[n * 16 + r * 4 + h] = sl;
            lj[n * 16 + r * 4 + h] = sr;
        }
}

// ------------------------------------------------- pass A: softmax (m, s) ---
// One thread per row i; lj[b][*][16] staged in 64KB LDS; single int4 stream
// over adj[b][i][*][0..3].  Online scalar softmax; all-masked row -> uniform.
__global__ __launch_bounds__(256) void k_ms(const int* __restrict__ adj,
                                            const float* __restrict__ li,
                                            const float* __restrict__ lj,
                                            float* __restrict__ m_out,
                                            float* __restrict__ is_out) {
    __shared__ float slj[1024 * 16];                       // 64 KB
    const int b = blockIdx.y;
    const int i = blockIdx.x * 256 + threadIdx.x;

    const float4* src = (const float4*)(lj + b * 1024 * 16);
    float4* dst = (float4*)slj;
    for (int t = threadIdx.x; t < 1024 * 16 / 4; t += 256) dst[t] = src[t];
    __syncthreads();

    float liv[16], m[16], s[16];
#pragma unroll
    for (int k = 0; k < 16; ++k) {
        liv[k] = li[(b * 1024 + i) * 16 + k];
        m[k] = GAT_NEG_INF;
        s[k] = 0.0f;
    }

    const int4* arow = (const int4*)(adj + (size_t)(b * 1024 + i) * 1024 * 4);
    for (int j = 0; j < 1024; ++j) {
        int4 av = arow[j];
        int ac[4] = {av.x, av.y, av.z, av.w};
#pragma unroll
        for (int r = 0; r < 4; ++r) {
            if (ac[r]) {
                float4 lv = ((const float4*)(slj + j * 16))[r];
                float lf[4] = {lv.x, lv.y, lv.z, lv.w};
#pragma unroll
                for (int hh = 0; hh < 4; ++hh) {
                    int k = r * 4 + hh;
                    float x = leaky(liv[k] + lf[hh]);
                    if (x > m[k]) { s[k] = s[k] * __expf(m[k] - x) + 1.0f; m[k] = x; }
                    else          { s[k] += __expf(x - m[k]); }
                }
            }
        }
    }
#pragma unroll
    for (int k = 0; k < 16; ++k) {
        if (s[k] == 0.0f) s[k] = 1024.0f;   // fully masked row -> uniform
        m_out[(b * 1024 + i) * 16 + k]  = m[k];
        is_out[(b * 1024 + i) * 16 + k] = 1.0f / s[k];
    }
}

// --------------------------------------- pass B: WMMA flash aggregation ----
// Block = (b, 16-row i-tile); 4 waves, wave = head h.  Adjacency read once as
// int4 (all 4 relations), staged via double-buffered ASYNC global->LDS loads
// overlapped with the WMMA/exp compute of the previous tile.
__global__ __launch_bounds__(128) void k_aggr(const int* __restrict__ adj,
                                              const __bf16* __restrict__ ft,
                                              const float* __restrict__ li,
                                              const float* __restrict__ lj,
                                              const float* __restrict__ m_in,
                                              const float* __restrict__ is_in,
                                              float* __restrict__ out) {
    __shared__ int4 sadj[2][16][32];                    // 16 KB: [i][j] x 4 rel
    __shared__ __align__(16) float slj[2][32][16];      //  4 KB: [j][r*4+h]

    const int b   = blockIdx.y;
    const int i0  = blockIdx.x * 16;
    const int tid = threadIdx.x;
    const int h   = tid >> 5;        // wave id == head
    const int l   = tid & 31;
    const int lr  = l & 15;
    const bool hih = l >= 16;

    // Per-lane row stats for A-fragment rows (A lane -> row l&15).
    float li4[4], m4[4], is4[4];
    const int rowb = (b * 1024 + i0 + lr) * 16;
#pragma unroll
    for (int r = 0; r < 4; ++r) {
        li4[r] = li[rowb + r * 4 + h];
        m4[r]  = m_in[rowb + r * 4 + h];
        is4[r] = is_in[rowb + r * 4 + h];
    }

    auto stage = [&](int buf, int j0) {
#pragma unroll
        for (int k = 0; k < 4; ++k) {                  // 512 int4 / 128 thr
            int flat = k * 128 + tid;
            int ii = flat >> 5, jj = flat & 31;
            const int* g =
                adj + (size_t)((b * 1024 + i0 + ii) * 1024 + j0 + jj) * 4;
#if GAT_ASYNC_LDS
            async_copy16(g, &sadj[buf][ii][jj]);
#else
            sadj[buf][ii][jj] = *(const int4*)g;
#endif
        }
        int jj = tid >> 2, part = tid & 3;
        const float* g2 = lj + (b * 1024 + j0 + jj) * 16 + part * 4;
#if GAT_ASYNC_LDS
        async_copy16(g2, &slj[buf][jj][part * 4]);
#else
        *(float4*)(&slj[buf][jj][part * 4]) = *(const float4*)g2;
        if (j0 + 32 < 1024)                            // global_prefetch_b8
            __builtin_prefetch(adj + (size_t)((b * 1024 + i0 + (tid >> 5)) * 1024
                                              + (j0 + 32) + (tid & 31)) * 4, 0, 0);
#endif
    };

    v8f acc[4] = {v8f{}, v8f{}, v8f{}, v8f{}};
    const __bf16* ftb = ft + ((b * 4 + h) << 16);
    const int o = hih ? 8 : 0;

    stage(0, 0);                                       // prologue tile
    for (int j0 = 0; j0 < 1024; j0 += 32) {
        const int cur = (j0 >> 5) & 1;
#if GAT_ASYNC_LDS
        __builtin_amdgcn_s_wait_asynccnt(0);           // own async ops done
#endif
        __syncthreads();                               // all waves' tiles done
        if (j0 + 32 < 1024) stage(cur ^ 1, j0 + 32);   // overlap next tile

#pragma unroll
        for (int r = 0; r < 4; ++r) {
            // A fragment: lane row = l&15; element e -> K = o + (e<8 ? e : e+8)
            Frag16 af;
#pragma unroll
            for (int e = 0; e < 16; ++e) {
                int jl = o + ((e < 8) ? e : (e + 8));
                int conn = ((const int*)&sadj[cur][lr][jl])[r];
                float x  = leaky(li4[r] + slj[cur][jl][r * 4 + h]);
                float sc = conn ? x : GAT_NEG_INF;
                float p  = __expf(sc - m4[r]) * is4[r];
                af.v[e] = (__bf16)p;
            }
            const __bf16* fb = ftb + (r << 21);        // + r*32*65536
            const int kb = j0 + (hih ? 8 : 0);
#pragma unroll
            for (int ct = 0; ct < 4; ++ct) {
                Frag16 bf;
                const __bf16* p = fb + (ct * 16 + lr) * 1024 + kb;
                bf.q[0] = *(const uint4*)(p);
                bf.q[1] = *(const uint4*)(p + 16);
                acc[ct] = __builtin_amdgcn_wmma_f32_16x16x32_bf16(
                    false, af.v, false, bf.v, (short)0, acc[ct], false, false);
            }
        }
    }

    // D layout: lane l, vgpr vi -> row (vi + (hih?8:0)), col (l&15)
#pragma unroll
    for (int ct = 0; ct < 4; ++ct)
#pragma unroll
        for (int vi = 0; vi < 8; ++vi) {
            int irow = i0 + vi + (hih ? 8 : 0);
            out[(b * 1024 + irow) * 256 + h * 64 + ct * 16 + lr] = acc[ct][vi];
        }
}

// ------------------------------------------------------------------ host ----
extern "C" void kernel_launch(void* const* d_in, const int* in_sizes, int n_in,
                              void* d_out, int out_size, void* d_ws, size_t ws_size,
                              hipStream_t stream) {
    (void)in_sizes; (void)n_in; (void)out_size; (void)ws_size;

    const float* nf  = (const float*)d_in[0];   // (8,1024,256) f32
    const int*   adj = (const int*)d_in[1];     // (8,1024,1024,4) i32
    const float* W   = (const float*)d_in[2];   // (4,256,256) f32
    const float* a   = (const float*)d_in[3];   // (4,4,128) f32
    float* out = (float*)d_out;                 // (8,1024,256) f32

    char* ws = (char*)d_ws;
    __bf16* nfb = (__bf16*)ws;                                  //  4 MiB
    __bf16* wt  = (__bf16*)(ws + (4u << 20));                   // 512 KiB
    __bf16* ft  = (__bf16*)(ws + (4u << 20) + (512u << 10));    // 16 MiB
    float*  li  = (float*)(ws + 21495808u);                     // 512 KiB
    float*  lj  = (float*)(ws + 22020096u);                     // 512 KiB
    float*  mb  = (float*)(ws + 22544384u);                     // 512 KiB
    float*  isb = (float*)(ws + 23068672u);                     // 512 KiB

    k_prep_nf<<<8192, 256, 0, stream>>>(nf, nfb);
    k_prep_w<<<1024, 256, 0, stream>>>(W, wt);

    dim3 g1(512, 4, 4);
    k_gemm_f<<<g1, 32, 0, stream>>>(nfb, wt, ft);

    k_lilj<<<32, 256, 0, stream>>>(ft, a, li, lj);

    dim3 g2(4, 8);
    k_ms<<<g2, 256, 0, stream>>>(adj, li, lj, mb, isb);

    dim3 g3(64, 8);
    k_aggr<<<g3, 128, 0, stream>>>(adj, ft, li, lj, mb, isb, out);
}
